// RRN_54855322305087
// MI455X (gfx1250) — compile-verified, matrix operand order checked by reading, not run on hardware
//
#include <hip/hip_runtime.h>
#include <hip/hip_bf16.h>

// ---------------------------------------------------------------------------
// RRN (recurrent relational net) on sudoku — CDNA5 / gfx1250 WMMA pipeline.
// GEMMs on V_WMMA_F32_16X16X32_F16; rel/g-layer weights staged to LDS via the
// Tensor Data Mover (tensor_load_to_lds, TENSORcnt), overlapped with gathers.
// ---------------------------------------------------------------------------

#define NNC   81
#define HID   96
#define BATCH 128
#define ROWS  (BATCH * NNC)   // 10368  (divisible by 16/32/64)
#define DEG   20
#define ITERS 4

typedef __attribute__((ext_vector_type(16))) _Float16 v16h;
typedef __attribute__((ext_vector_type(8)))  _Float16 v8h;
typedef __attribute__((ext_vector_type(8)))  float    v8f;
typedef __attribute__((ext_vector_type(4)))  unsigned int u32x4;
typedef __attribute__((ext_vector_type(4)))  int      i32x4;
typedef __attribute__((ext_vector_type(8)))  int      i32x8;

#if defined(__HIP_DEVICE_COMPILE__) && defined(__gfx1250__) &&       \
    __has_builtin(__builtin_amdgcn_tensor_load_to_lds) &&            \
    __has_builtin(__builtin_amdgcn_s_wait_tensorcnt)
#define USE_TDM 1
#warning "CDNA5 probe: TDM path ENABLED (tensor_load_to_lds live in device pass)"
#else
#define USE_TDM 0
#if defined(__HIP_DEVICE_COMPILE__)
#warning "CDNA5 probe: TDM path DISABLED in device pass (fallback LDS copy)"
#endif
#endif

// ---- WMMA fragment helpers -------------------------------------------------
// 16-bit A-matrix 16x32 layout (ISA 7.12.2): lane l<16 holds row M=l,
// halfs 0..7 = K 0..7, halfs 8..15 = K 16..23; lane l+16 holds K+8 chunks.
// Row-major NxK weights (x @ W.T) use the identical layout for the B operand.
__device__ __forceinline__ v16h frag_load(const _Float16* p, int stride,
                                          int row0, int k0, int lane) {
  const _Float16* base =
      p + (size_t)(row0 + (lane & 15)) * stride + k0 + ((lane & 16) ? 8 : 0);
  v8h lo = *(const v8h*)(base);
  v8h hi = *(const v8h*)(base + 16);
  return __builtin_shufflevector(lo, hi, 0, 1, 2, 3, 4, 5, 6, 7,
                                 8, 9, 10, 11, 12, 13, 14, 15);
}

template <int KS>
__device__ __forceinline__ v8f tile_gemm(const _Float16* A, int lda,
                                         const _Float16* W, int ldw,
                                         int mt, int nt, int lane, v8f acc) {
#pragma unroll
  for (int k = 0; k < KS; ++k) {
    v16h a = frag_load(A, lda, mt * 16, k * 32, lane);
    v16h b = frag_load(W, ldw, nt * 16, k * 32, lane);
    acc = __builtin_amdgcn_wmma_f32_16x16x32_f16(false, a, false, b,
                                                 (short)0, acc, false, false);
  }
  return acc;
}

// C/D 16x16 f32 tile: VGPR r -> row mt*16 + r + (lane&16 ? 8:0), col nt*16+(lane&15)
__device__ __forceinline__ void tile_store_f16(v8f acc, _Float16* O, int ldo,
                                               int mt, int nt, const float* bias,
                                               bool relu, int lane) {
  int col = nt * 16 + (lane & 15);
  float bv = bias ? bias[col] : 0.f;
  int r0 = mt * 16 + ((lane & 16) ? 8 : 0);
#pragma unroll
  for (int r = 0; r < 8; ++r) {
    float v = acc[r] + bv;
    if (relu) v = fmaxf(v, 0.f);
    O[(size_t)(r0 + r) * ldo + col] = (_Float16)v;
  }
}

__device__ __forceinline__ void tile_store_f32(v8f acc, float* O, int ldo,
                                               int mt, int nt, const float* bias,
                                               int lane) {
  int col = nt * 16 + (lane & 15);
  float bv = bias ? bias[col] : 0.f;
  int r0 = mt * 16 + ((lane & 16) ? 8 : 0);
#pragma unroll
  for (int r = 0; r < 8; ++r) O[(size_t)(r0 + r) * ldo + col] = acc[r] + bv;
}

// ---- TDM: 1-D tile (n8 x 8-byte elements) global -> LDS --------------------
// D# per ISA 8.3/8.4: group0 = {count=1, lds_addr, global_addr, type=2},
// group1 = {data_size=8B, tensor_dim0=tile_dim0=n8, dims1 collapsed to 1}.
#if USE_TDM
__device__ __forceinline__ void tdm_stage(const void* gsrc,
                                          unsigned lds_byte_off, unsigned n8) {
  unsigned long long ga = (unsigned long long)(size_t)gsrc;
  u32x4 g0 = {1u, lds_byte_off, (unsigned)ga,
              (unsigned)((ga >> 32) & 0x1FFFFFFULL) | (2u << 30)};
  i32x8 g1 = {(int)(3u << 16),                              // data_size = 8B
              (int)((n8 & 0xffffu) << 16),                  // tensor_dim0 lo16
              (int)(((n8 >> 16) & 0xffffu) | (1u << 16)),   // dim0 hi16 | tensor_dim1=1
              (int)((n8 & 0xffffu) << 16),                  // tile_dim0
              1,                                            // tile_dim1 = 1
              (int)n8,                                      // dim0_stride lo32
              (int)((n8 & 0xffffu) << 16),                  // | dim1_stride lo16
              (int)(n8 >> 16)};                             // dim1_stride hi
  i32x4 z4 = {0, 0, 0, 0};
#if __clang_major__ >= 23
  i32x8 z8 = {0, 0, 0, 0, 0, 0, 0, 0};
  __builtin_amdgcn_tensor_load_to_lds(g0, g1, z4, z4, z8, 0);
#else
  __builtin_amdgcn_tensor_load_to_lds(g0, g1, z4, z4, 0);
#endif
}
#endif

#define WALL_BYTES ((96 * 192 + 2 * 96 * 96) * 2)  // 73728 B, 16B-vector count
#define WALL_V8H   (WALL_BYTES / 16)

// ---- weight f32 -> f16 with zero padding (rows x cols -> drows x dcols) ----
__global__ __launch_bounds__(256) void k_cvt(const float* __restrict__ src,
                                             _Float16* __restrict__ dst,
                                             int rows, int cols, int drows,
                                             int dcols) {
  int i = blockIdx.x * 256 + threadIdx.x;
  int total = drows * dcols;
  if (i >= total) return;
  int r = i / dcols, c = i % dcols;
  float v = (r < rows && c < cols) ? src[r * cols + c] : 0.f;
  dst[i] = (_Float16)v;
}

// ---- embed + in_mlp: X = mlp(embed[grids]); 64 rows / block ---------------
__global__ __launch_bounds__(256) void k_in_mlp(
    const int* __restrict__ grids, const float* __restrict__ embed,
    const _Float16* __restrict__ W0, const float* __restrict__ b0,
    const _Float16* __restrict__ W1, const float* __restrict__ b1,
    const _Float16* __restrict__ W2, const float* __restrict__ b2,
    _Float16* __restrict__ X) {
  extern __shared__ __align__(16) char smem[];
  _Float16* sA = (_Float16*)smem;     // 64 x 32 (K padded 16->32)
  _Float16* s1 = sA + 64 * 32;        // 64 x 96
  _Float16* s2 = s1 + 64 * 96;        // 64 x 96
  int tid = threadIdx.x, lane = tid & 31, wave = tid >> 5;
  int base = blockIdx.x * 64;

  __builtin_prefetch(W1, 0, 3);
  __builtin_prefetch(W2, 0, 3);
  for (int idx = tid; idx < 64 * 32; idx += 256) {
    int r = idx >> 5, c = idx & 31;
    int g = grids[base + r];
    sA[idx] = (c < 16) ? (_Float16)embed[g * 16 + c] : (_Float16)0.f;
  }
  __syncthreads();
  for (int t = wave; t < 24; t += 8) {
    int mt = t / 6, nt = t % 6;
    v8f acc = {};
    acc = tile_gemm<1>(sA, 32, W0, 32, mt, nt, lane, acc);
    tile_store_f16(acc, s1, 96, mt, nt, b0, true, lane);
  }
  __syncthreads();
  for (int t = wave; t < 24; t += 8) {
    int mt = t / 6, nt = t % 6;
    v8f acc = {};
    acc = tile_gemm<3>(s1, 96, W1, 96, mt, nt, lane, acc);
    tile_store_f16(acc, s2, 96, mt, nt, b1, true, lane);
  }
  __syncthreads();
  for (int t = wave; t < 24; t += 8) {
    int mt = t / 6, nt = t % 6;
    v8f acc = {};
    acc = tile_gemm<3>(s2, 96, W2, 96, mt, nt, lane, acc);
    tile_store_f16(acc, X + (size_t)base * HID, HID, mt, nt, b2, false, lane);
  }
}

// ---- rel layer: 1 block = 4 cells x 20 edges = 80 rows (5 M-tiles) --------
// Weights (W0|W1|W2 contiguous, 73728 B) staged to LDS by the TDM while the
// neighbor gather runs; all GEMM operands then come from LDS.
__global__ __launch_bounds__(256) void k_rel(
    const _Float16* __restrict__ H, const int* __restrict__ edges,
    const _Float16* __restrict__ Wall, const float* __restrict__ b0,
    const float* __restrict__ b1, const float* __restrict__ b2,
    _Float16* __restrict__ M) {
  extern __shared__ __align__(16) char smem[];
  _Float16* sW0 = (_Float16*)smem;         // 96 x 192
  _Float16* sW1 = sW0 + 96 * 192;          // 96 x 96
  _Float16* sW2 = sW1 + 96 * 96;           // 96 x 96
  _Float16* sA = sW2 + 96 * 96;            // 80 x 192 (aliased by sF below)
  float* sF = (float*)sA;                  // 80 x 96 f32 (layer-3 pre-sum)
  _Float16* s1 = sA + 80 * 192;            // 80 x 96
  _Float16* s2 = s1 + 80 * 96;             // 80 x 96
  int* sSelf = (int*)(s2 + 80 * 96);       // 80
  int* sNbr = sSelf + 80;                  // 80
  int tid = threadIdx.x, lane = tid & 31, wave = tid >> 5;
  int cell0 = blockIdx.x * 4;

#if USE_TDM
  if (wave == 0) {
    unsigned ldsW = (unsigned)__builtin_amdgcn_groupstaticsize();  // dyn base
    tdm_stage(Wall, ldsW, WALL_BYTES / 8);                         // 9216 x 8B
  }
#else
  for (int c = tid; c < WALL_V8H; c += 256)
    ((v8h*)sW0)[c] = ((const v8h*)Wall)[c];
#endif

  if (tid < 80) {
    int cell = cell0 + tid / DEG;
    int b = cell / NNC, i = cell % NNC;
    sSelf[tid] = cell;
    sNbr[tid] = b * NNC + edges[i * DEG + (tid % DEG)];
  }
  __syncthreads();
  // gather A = [H_self | H_nbr], 8-half vector chunks (overlaps the TDM DMA)
  for (int c = tid; c < 80 * 24; c += 256) {
    int r = c / 24, o = (c % 24) * 8;
    const _Float16* src = (o < HID)
                              ? (H + (size_t)sSelf[r] * HID + o)
                              : (H + (size_t)sNbr[r] * HID + (o - HID));
    *(v8h*)(sA + r * 192 + o) = *(const v8h*)src;
  }
#if USE_TDM
  if (wave == 0) __builtin_amdgcn_s_wait_tensorcnt(0);
#endif
  __syncthreads();
  for (int t = wave; t < 30; t += 8) {
    int mt = t / 6, nt = t % 6;
    v8f acc = {};
    acc = tile_gemm<6>(sA, 192, sW0, 192, mt, nt, lane, acc);
    tile_store_f16(acc, s1, 96, mt, nt, b0, true, lane);
  }
  __syncthreads();
  for (int t = wave; t < 30; t += 8) {
    int mt = t / 6, nt = t % 6;
    v8f acc = {};
    acc = tile_gemm<3>(s1, 96, sW1, 96, mt, nt, lane, acc);
    tile_store_f16(acc, s2, 96, mt, nt, b1, true, lane);
  }
  __syncthreads();
  for (int t = wave; t < 30; t += 8) {  // sA is dead; sF may alias it
    int mt = t / 6, nt = t % 6;
    v8f acc = {};
    acc = tile_gemm<3>(s2, 96, sW2, 96, mt, nt, lane, acc);
    tile_store_f32(acc, sF, 96, mt, nt, b2, lane);
  }
  __syncthreads();
  // per-cell sum over 20 messages -> M (f16)
  for (int o = tid; o < 4 * HID; o += 256) {
    int c = o / HID, col = o % HID;
    const float* p = sF + (size_t)(c * DEG) * 96 + col;
    float s = 0.f;
#pragma unroll
    for (int e = 0; e < DEG; ++e) s += p[e * 96];
    M[(size_t)(cell0 + c) * HID + col] = (_Float16)s;
  }
}

// ---- g_mlp: x_in = mlp([X | M]); 64 rows / block; weights TDM-staged ------
__global__ __launch_bounds__(256) void k_gmlp(
    const _Float16* __restrict__ X, const _Float16* __restrict__ M,
    const _Float16* __restrict__ Wall, const float* __restrict__ b0,
    const float* __restrict__ b1, const float* __restrict__ b2,
    _Float16* __restrict__ XIN) {
  extern __shared__ __align__(16) char smem[];
  _Float16* sW0 = (_Float16*)smem;   // 96 x 192
  _Float16* sW1 = sW0 + 96 * 192;    // 96 x 96
  _Float16* sW2 = sW1 + 96 * 96;     // 96 x 96
  _Float16* sA = sW2 + 96 * 96;      // 64 x 192
  _Float16* s1 = sA + 64 * 192;      // 64 x 96
  _Float16* s2 = s1 + 64 * 96;       // 64 x 96
  int tid = threadIdx.x, lane = tid & 31, wave = tid >> 5;
  int base = blockIdx.x * 64;

#if USE_TDM
  if (wave == 0) {
    unsigned ldsW = (unsigned)__builtin_amdgcn_groupstaticsize();
    tdm_stage(Wall, ldsW, WALL_BYTES / 8);
  }
#else
  for (int c = tid; c < WALL_V8H; c += 256)
    ((v8h*)sW0)[c] = ((const v8h*)Wall)[c];
#endif
  for (int c = tid; c < 64 * 24; c += 256) {
    int r = c / 24, o = (c % 24) * 8;
    const _Float16* src = (o < HID) ? (X + (size_t)(base + r) * HID + o)
                                    : (M + (size_t)(base + r) * HID + (o - HID));
    *(v8h*)(sA + r * 192 + o) = *(const v8h*)src;
  }
#if USE_TDM
  if (wave == 0) __builtin_amdgcn_s_wait_tensorcnt(0);
#endif
  __syncthreads();
  for (int t = wave; t < 24; t += 8) {
    int mt = t / 6, nt = t % 6;
    v8f acc = {};
    acc = tile_gemm<6>(sA, 192, sW0, 192, mt, nt, lane, acc);
    tile_store_f16(acc, s1, 96, mt, nt, b0, true, lane);
  }
  __syncthreads();
  for (int t = wave; t < 24; t += 8) {
    int mt = t / 6, nt = t % 6;
    v8f acc = {};
    acc = tile_gemm<3>(s1, 96, sW1, 96, mt, nt, lane, acc);
    tile_store_f16(acc, s2, 96, mt, nt, b1, true, lane);
  }
  __syncthreads();
  for (int t = wave; t < 24; t += 8) {
    int mt = t / 6, nt = t % 6;
    v8f acc = {};
    acc = tile_gemm<3>(s2, 96, sW2, 96, mt, nt, lane, acc);
    tile_store_f16(acc, XIN + (size_t)base * HID, HID, mt, nt, b2, false, lane);
  }
}

// ---- LSTM cell: gates = x_in@Wih.T + h@Whh.T; 32 rows / block -------------
__global__ __launch_bounds__(256) void k_lstm(
    const _Float16* __restrict__ XIN, _Float16* __restrict__ H,
    float* __restrict__ C, const _Float16* __restrict__ Wih,
    const _Float16* __restrict__ Whh, const float* __restrict__ bih,
    const float* __restrict__ bhh) {
  extern __shared__ __align__(16) char smem[];
  _Float16* sX = (_Float16*)smem;      // 32 x 96
  _Float16* sH = sX + 32 * 96;         // 32 x 96
  float* sG = (float*)(sH + 32 * 96);  // 32 x 384 (gates, f32)
  int tid = threadIdx.x, lane = tid & 31, wave = tid >> 5;
  int base = blockIdx.x * 32;

  __builtin_prefetch(Wih, 0, 3);
  __builtin_prefetch(Whh, 0, 3);
  for (int c = tid; c < 32 * 12 * 2; c += 256) {
    int half = c >= 384;
    int cc = c - half * 384;
    int r = cc / 12, o = (cc % 12) * 8;
    if (half)
      *(v8h*)(sH + r * 96 + o) = *(const v8h*)(H + (size_t)(base + r) * HID + o);
    else
      *(v8h*)(sX + r * 96 + o) = *(const v8h*)(XIN + (size_t)(base + r) * HID + o);
  }
  __syncthreads();
  // 2 M-tiles x 24 N-tiles, both GEMMs accumulated into one tile
  for (int t = wave; t < 48; t += 8) {
    int mt = t / 24, nt = t % 24;
    v8f acc = {};
    acc = tile_gemm<3>(sX, 96, Wih, 96, mt, nt, lane, acc);
    acc = tile_gemm<3>(sH, 96, Whh, 96, mt, nt, lane, acc);
    tile_store_f32(acc, sG, 384, mt, nt, nullptr, lane);
  }
  __syncthreads();
  for (int o = tid; o < 32 * HID; o += 256) {
    int r = o / HID, col = o % HID;
    size_t row = (size_t)(base + r);
    float gi = sG[r * 384 + col] + bih[col] + bhh[col];
    float gf = sG[r * 384 + 96 + col] + bih[96 + col] + bhh[96 + col];
    float gg = sG[r * 384 + 192 + col] + bih[192 + col] + bhh[192 + col];
    float go = sG[r * 384 + 288 + col] + bih[288 + col] + bhh[288 + col];
    float si = 1.f / (1.f + __expf(-gi));
    float sf = 1.f / (1.f + __expf(-gf));
    float so = 1.f / (1.f + __expf(-go));
    float c2 = sf * C[row * HID + col] + si * tanhf(gg);
    float h2 = so * tanhf(c2);
    C[row * HID + col] = c2;
    H[row * HID + col] = (_Float16)h2;  // h2 == next H (same buffer)
  }
}

// ---- r_mlp: out = mlp(H) -> (rows x 9) f32; 64 rows / block ---------------
__global__ __launch_bounds__(256) void k_rmlp(
    const _Float16* __restrict__ H, const _Float16* __restrict__ W0,
    const float* __restrict__ b0, const _Float16* __restrict__ W1,
    const float* __restrict__ b1, const _Float16* __restrict__ W2 /*16x96*/,
    const float* __restrict__ b2 /*9*/, float* __restrict__ OUT) {
  extern __shared__ __align__(16) char smem[];
  _Float16* sA = (_Float16*)smem;  // 64 x 96
  _Float16* s1 = sA + 64 * 96;     // 64 x 96
  _Float16* s2 = s1 + 64 * 96;     // 64 x 96
  int tid = threadIdx.x, lane = tid & 31, wave = tid >> 5;
  int base = blockIdx.x * 64;

  __builtin_prefetch(W0, 0, 3);
  __builtin_prefetch(W1, 0, 3);
  __builtin_prefetch(W2, 0, 3);
  for (int c = tid; c < 64 * 12; c += 256) {
    int r = c / 12, o = (c % 12) * 8;
    *(v8h*)(sA + r * 96 + o) = *(const v8h*)(H + (size_t)(base + r) * HID + o);
  }
  __syncthreads();
  for (int t = wave; t < 24; t += 8) {
    int mt = t / 6, nt = t % 6;
    v8f acc = {};
    acc = tile_gemm<3>(sA, 96, W0, 96, mt, nt, lane, acc);
    tile_store_f16(acc, s1, 96, mt, nt, b0, true, lane);
  }
  __syncthreads();
  for (int t = wave; t < 24; t += 8) {
    int mt = t / 6, nt = t % 6;
    v8f acc = {};
    acc = tile_gemm<3>(s1, 96, W1, 96, mt, nt, lane, acc);
    tile_store_f16(acc, s2, 96, mt, nt, b1, true, lane);
  }
  __syncthreads();
  // final layer: N=9 (weight rows padded to 16), single N-tile
  for (int t = wave; t < 4; t += 8) {
    v8f acc = {};
    acc = tile_gemm<3>(s2, 96, W2, 96, t, 0, lane, acc);
    int col = lane & 15;
    int r0 = t * 16 + ((lane & 16) ? 8 : 0);
    if (col < 9) {
      float bv = b2[col];
#pragma unroll
      for (int r = 0; r < 8; ++r)
        OUT[(size_t)(base + r0 + r) * 9 + col] = acc[r] + bv;
    }
  }
}

// ---------------------------------------------------------------------------
extern "C" void kernel_launch(void* const* d_in, const int* in_sizes, int n_in,
                              void* d_out, int out_size, void* d_ws,
                              size_t ws_size, hipStream_t stream) {
  const int* grids = (const int*)d_in[0];
  const float* c0 = (const float*)d_in[2];  // (1, B*NN, HID) f32
  const int* edges = (const int*)d_in[3];
  const float* embed = (const float*)d_in[4];

  const float *inW[3], *inB[3], *relW[3], *relB[3], *gW[3], *gB[3], *rW[3],
      *rB[3];
  const float *Wih, *Whh, *bih, *bhh;
  int p = 5;
  auto take_mlp = [&](const float** W, const float** Bv) {
    for (int i = 0; i < 3; ++i) {
      W[i] = (const float*)d_in[p++];
      Bv[i] = (const float*)d_in[p++];
    }
  };
  auto take_lstm = [&]() {
    Wih = (const float*)d_in[p++];
    Whh = (const float*)d_in[p++];
    bih = (const float*)d_in[p++];
    bhh = (const float*)d_in[p++];
  };
  if (in_sizes[5] == 96 * 16) {
    // dict insertion order: embed, in_mlp, rel_mlp, g_mlp, lstm, r_mlp
    take_mlp(inW, inB);
    take_mlp(relW, relB);
    take_mlp(gW, gB);
    take_lstm();
    take_mlp(rW, rB);
  } else {
    // sorted-key pytree order: embed, g_mlp, in_mlp, lstm, r_mlp, rel_mlp
    take_mlp(gW, gB);
    take_mlp(inW, inB);
    take_lstm();
    take_mlp(rW, rB);
    take_mlp(relW, relB);
  }

  // ---- workspace carve ----
  size_t off = 0;
  auto alloc = [&](size_t bytes) {
    off = (off + 255) & ~(size_t)255;
    void* r = (char*)d_ws + off;
    off += bytes;
    return r;
  };
  _Float16* w_in0 = (_Float16*)alloc(96 * 32 * 2);
  _Float16* w_in1 = (_Float16*)alloc(96 * 96 * 2);
  _Float16* w_in2 = (_Float16*)alloc(96 * 96 * 2);
  // rel / g weights each one contiguous block (single TDM descriptor each)
  _Float16* w_relAll = (_Float16*)alloc(WALL_BYTES);
  _Float16* w_rel0 = w_relAll;
  _Float16* w_rel1 = w_rel0 + 96 * 192;
  _Float16* w_rel2 = w_rel1 + 96 * 96;
  _Float16* w_gAll = (_Float16*)alloc(WALL_BYTES);
  _Float16* w_g0 = w_gAll;
  _Float16* w_g1 = w_g0 + 96 * 192;
  _Float16* w_g2 = w_g1 + 96 * 96;
  _Float16* w_ih = (_Float16*)alloc(384 * 96 * 2);
  _Float16* w_hh = (_Float16*)alloc(384 * 96 * 2);
  _Float16* w_r0 = (_Float16*)alloc(96 * 96 * 2);
  _Float16* w_r1 = (_Float16*)alloc(96 * 96 * 2);
  _Float16* w_r2 = (_Float16*)alloc(16 * 96 * 2);
  _Float16* Xh = (_Float16*)alloc((size_t)ROWS * HID * 2);
  _Float16* Hh = (_Float16*)alloc((size_t)ROWS * HID * 2);
  _Float16* Mh = (_Float16*)alloc((size_t)ROWS * HID * 2);
  _Float16* XINh = (_Float16*)alloc((size_t)ROWS * HID * 2);
  float* Cf = (float*)alloc((size_t)ROWS * HID * 4);

  // ---- weight conversion (f32 -> f16, padded) ----
  auto cvt = [&](const float* src, _Float16* dst, int rows, int cols, int drows,
                 int dcols) {
    int total = drows * dcols;
    k_cvt<<<(total + 255) / 256, 256, 0, stream>>>(src, dst, rows, cols, drows,
                                                   dcols);
  };
  cvt(inW[0], w_in0, 96, 16, 96, 32);
  cvt(inW[1], w_in1, 96, 96, 96, 96);
  cvt(inW[2], w_in2, 96, 96, 96, 96);
  cvt(relW[0], w_rel0, 96, 192, 96, 192);
  cvt(relW[1], w_rel1, 96, 96, 96, 96);
  cvt(relW[2], w_rel2, 96, 96, 96, 96);
  cvt(gW[0], w_g0, 96, 192, 96, 192);
  cvt(gW[1], w_g1, 96, 96, 96, 96);
  cvt(gW[2], w_g2, 96, 96, 96, 96);
  cvt(Wih, w_ih, 384, 96, 384, 96);
  cvt(Whh, w_hh, 384, 96, 384, 96);
  cvt(rW[0], w_r0, 96, 96, 96, 96);
  cvt(rW[1], w_r1, 96, 96, 96, 96);
  cvt(rW[2], w_r2, 9, 96, 16, 96);

  // c0 -> working C buffer (f32, same layout)
  hipMemcpyAsync(Cf, c0, (size_t)ROWS * HID * 4, hipMemcpyDeviceToDevice,
                 stream);

  // ---- X = in_mlp(embed[grids]); H0 = h0 = X ----
  size_t smem_in = 64 * 32 * 2 + 2 * (64 * 96 * 2);
  k_in_mlp<<<ROWS / 64, 256, smem_in, stream>>>(grids, embed, w_in0, inB[0],
                                                w_in1, inB[1], w_in2, inB[2],
                                                Xh);
  hipMemcpyAsync(Hh, Xh, (size_t)ROWS * HID * 2, hipMemcpyDeviceToDevice,
                 stream);

  size_t smem_rel = WALL_BYTES                     // staged weights
                    + 80 * 192 * 2                 // A (aliased by f32 pre-sum)
                    + 2 * (80 * 96 * 2)            // act1/act2
                    + 2 * 80 * 4;                  // idx arrays
  size_t smem_g = WALL_BYTES + 64 * 192 * 2 + 2 * (64 * 96 * 2);
  size_t smem_lstm = 2 * (32 * 96 * 2) + 32 * 384 * 4;
  size_t smem_r = 3 * (64 * 96 * 2);

  for (int it = 0; it < ITERS; ++it) {
    k_rel<<<ROWS / 4, 256, smem_rel, stream>>>(Hh, edges, w_relAll, relB[0],
                                               relB[1], relB[2], Mh);
    k_gmlp<<<ROWS / 64, 256, smem_g, stream>>>(Xh, Mh, w_gAll, gB[0], gB[1],
                                               gB[2], XINh);
    k_lstm<<<ROWS / 32, 256, smem_lstm, stream>>>(XINh, Hh, Cf, w_ih, w_hh,
                                                  bih, bhh);
    k_rmlp<<<ROWS / 64, 256, smem_r, stream>>>(
        Hh, w_r0, rB[0], w_r1, rB[1], w_r2, rB[2],
        (float*)d_out + (size_t)it * ROWS * 9);
  }
}
